// Encodelayer_4226247819316
// MI455X (gfx1250) — matrix operand order repaired and checked by reading, not compiled, not run on hardware
//
#include <hip/hip_runtime.h>
#include <hip/hip_bf16.h>
#include <cstdint>

// ---------------------------------------------------------------------------
// Transformer encoder layer for MI455X (gfx1250, wave32, WMMA).
// - All GEMMs: bf16 WMMA (v_wmma_f32_16x16x32_bf16), fp32 accumulate.
// - Global->LDS staging via async DMA (global_load_async_to_lds_b128),
//   double-buffered LDS, s_wait_asynccnt pipelining.
// - 64x64 register tile per wave (4x4 WMMA) for high math/LDS-byte ratio.
// - XOR-swizzled LDS (no padding): chunk ^= row>>2 kills the 64B-stride
//   bank aliasing between lanes 0/4/8/12 while keeping 16B contiguity.
// Problem: B=4, S=1024, D=1024, H=16, DK=64, DFF=4096. ~120 GFLOP; working
// set fits in 192MB L2 -> compute bound -> everything on the matrix pipes.
// ---------------------------------------------------------------------------

#define B_   4
#define S_   1024
#define D_   1024
#define H_   16
#define DK_  64
#define DFF_ 4096
#define EPS_ 1e-6f

typedef __attribute__((ext_vector_type(16))) __bf16 v16bf;
typedef __attribute__((ext_vector_type(8)))  float  v8f;

// Async global->LDS 16B copy. LDS address = low 32 bits of the flat pointer
// (flat->LDS mapping truncates to addr[31:0] per ISA 10.2).
__device__ __forceinline__ void async_copy16(void* lds_ptr, const void* gptr) {
  unsigned lds = (unsigned)(uintptr_t)lds_ptr;
  unsigned long long ga = (unsigned long long)(uintptr_t)gptr;
  asm volatile("global_load_async_to_lds_b128 %0, %1, off"
               :: "v"(lds), "v"(ga) : "memory");
}
__device__ __forceinline__ void wait_async0() {
  asm volatile("s_wait_asynccnt 0x0" ::: "memory");
}

// DPP ROW_ROR rotate-allreduce within 16-lane rows (VALU-only, no LDS).
#define DPP_ROR_F(x, n)                                              \
  __int_as_float(__builtin_amdgcn_update_dpp(                        \
      0, __float_as_int(x), 0x120 + (n), 0xF, 0xF, true))

__device__ __forceinline__ float rowmax16(float x) {
  x = fmaxf(x, DPP_ROR_F(x, 1));
  x = fmaxf(x, DPP_ROR_F(x, 2));
  x = fmaxf(x, DPP_ROR_F(x, 4));
  x = fmaxf(x, DPP_ROR_F(x, 8));
  return x;
}
__device__ __forceinline__ float rowsum16(float x) {
  x += DPP_ROR_F(x, 1);
  x += DPP_ROR_F(x, 2);
  x += DPP_ROR_F(x, 4);
  x += DPP_ROR_F(x, 8);
  return x;
}

// ---------------------------------------------------------------------------
// fp32 -> bf16 converts (straight, and transposed for weights)
// ---------------------------------------------------------------------------
__global__ void f32_to_bf16_kernel(const float* __restrict__ in,
                                   __bf16* __restrict__ out, int n) {
  int i = blockIdx.x * blockDim.x + threadIdx.x;
  int stride = gridDim.x * blockDim.x;
  for (; i < n; i += stride) out[i] = (__bf16)in[i];
}

// in: [K][N] fp32 row-major; out: [N][K] bf16 row-major (transposed)
__global__ void f32_to_bf16_T_kernel(const float* __restrict__ in,
                                     __bf16* __restrict__ out, int K, int N) {
  int i = blockIdx.x * blockDim.x + threadIdx.x;
  int stride = gridDim.x * blockDim.x;
  const int total = K * N;
  for (; i < total; i += stride) {
    const int n = i / K;
    const int k = i - n * K;
    out[i] = (__bf16)in[(size_t)k * N + n];
  }
}

// ---------------------------------------------------------------------------
// Tiled WMMA GEMM: C[M,N] = A[M,K](bf16) @ W[K,N] with W given transposed
// (WT[N][K], bf16). Compile-time epilogue: +bias, +resid, relu, fp32 out,
// bf16 out, bf16 transposed out ([N][M], for attention V).
// Block: 256 threads = 8 waves (2 M-slabs x 4 N-slabs); block tile 128x256,
// wave tile 64x64 (4x4 WMMA), K-step 32, async double-buffered swizzled LDS.
// ---------------------------------------------------------------------------
#define BM 128
#define BN 256
#define BK 32

// Swizzled element index within a 32-elem LDS row: 16B chunk ^= (row>>2)&3.
__device__ __forceinline__ int swz(int row, int k) {
  return (k & 7) | ((((k >> 3) ^ (row >> 2)) & 3) << 3);
}

template <bool RELU, bool RESID, bool OUTF, bool OUTB, bool OUTBT>
__global__ __launch_bounds__(256) void gemm_bf16_wmma(
    const __bf16* __restrict__ A, const __bf16* __restrict__ WT,
    const float* __restrict__ bias, const float* __restrict__ resid,
    float* __restrict__ outF, __bf16* __restrict__ outB,
    __bf16* __restrict__ outBT, int M, int N, int K) {
  __shared__ __bf16 As[2][BM][BK];  // [m][k], swizzled chunks (16KB)
  __shared__ __bf16 Ws[2][BN][BK];  // [n][k], swizzled chunks (32KB)

  const int tid  = threadIdx.x;
  const int lane = tid & 31;
  const int wave = tid >> 5;
  const int wn   = wave & 3;   // 4 column slabs of 64
  const int wmr  = wave >> 2;  // 2 row slabs of 64
  const int half = lane >> 4;
  const int lr   = lane & 15;

  const int m0 = blockIdx.y * BM;
  const int n0 = blockIdx.x * BN;

  // Stage one K-step tile pair: A 2 chunks/thread, W 4 chunks/thread (async).
  const int arow = tid >> 1;
  const int acb  = (tid & 1) * 2;  // chunk base (2 chunks of 8 elems)
  auto stage = [&](int buf, int k0) {
#pragma unroll
    for (int c = 0; c < 2; ++c) {
      const int ch = acb + c;
      const int sc = ch ^ ((arow >> 2) & 3);
      async_copy16(&As[buf][arow][sc * 8],
                   &A[(size_t)(m0 + arow) * K + k0 + ch * 8]);
    }
#pragma unroll
    for (int c = 0; c < 4; ++c) {
      const int sc = c ^ ((tid >> 2) & 3);
      async_copy16(&Ws[buf][tid][sc * 8],
                   &WT[(size_t)(n0 + tid) * K + k0 + c * 8]);
    }
  };

  v8f acc[4][4];
#pragma unroll
  for (int mt = 0; mt < 4; ++mt)
#pragma unroll
    for (int nt = 0; nt < 4; ++nt) acc[mt][nt] = (v8f)0.0f;

  const int nk = K / BK;
  stage(0, 0);
  for (int kt = 0; kt < nk; ++kt) {
    const int buf = kt & 1;
    wait_async0();       // our tile-kt DMA landed
    __syncthreads();     // everyone's tile-kt data visible
    if (kt + 1 < nk) stage(buf ^ 1, (kt + 1) * BK);  // overlap next tile

    // Load all four B fragments (held across the mt loop).
    v16bf bfr[4];
#pragma unroll
    for (int nt = 0; nt < 4; ++nt) {
      const int col = wn * 64 + nt * 16 + lr;
#pragma unroll
      for (int v = 0; v < 8; ++v) {
        const int k = ((v < 4) ? 2 * v : 16 + 2 * (v - 4)) + 8 * half;  // even -> pair in one chunk
        const int idx = swz(col, k);
        bfr[nt][2 * v]     = Ws[buf][col][idx];
        bfr[nt][2 * v + 1] = Ws[buf][col][idx + 1];
      }
    }
#pragma unroll
    for (int mt = 0; mt < 4; ++mt) {
      v16bf afr;
      const int row = wmr * 64 + mt * 16 + lr;
#pragma unroll
      for (int v = 0; v < 8; ++v) {
        const int kb = ((v < 4) ? 2 * v : 16 + 2 * (v - 4)) + 8 * half;
        const int idx = swz(row, kb);
        afr[2 * v]     = As[buf][row][idx];
        afr[2 * v + 1] = As[buf][row][idx + 1];
      }
#pragma unroll
      for (int nt = 0; nt < 4; ++nt) {
        acc[mt][nt] = __builtin_amdgcn_wmma_f32_16x16x32_bf16(
            false, afr, false, bfr[nt], (short)0, acc[mt][nt], false, false);
      }
    }
    __syncthreads();  // done reading buf before it is overwritten
  }

  // Branch-free epilogue. C layout: lane -> col, VGPR v -> row v + 8*half.
#pragma unroll
  for (int mt = 0; mt < 4; ++mt) {
#pragma unroll
    for (int nt = 0; nt < 4; ++nt) {
      const int col = n0 + wn * 64 + nt * 16 + lr;
      const float bc = bias[col];
#pragma unroll
      for (int v = 0; v < 8; ++v) {
        const int row = m0 + wmr * 64 + mt * 16 + v + 8 * half;
        const size_t idx = (size_t)row * N + col;
        float val = acc[mt][nt][v] + bc;
        if constexpr (RESID) val += resid[idx];
        if constexpr (RELU)  val = fmaxf(val, 0.0f);
        if constexpr (OUTF)  outF[idx] = val;
        if constexpr (OUTB)  outB[idx] = (__bf16)val;
        if constexpr (OUTBT) outBT[(size_t)col * M + row] = (__bf16)val;
      }
    }
  }
}

// ---------------------------------------------------------------------------
// Flash attention: softmax(Q Kᵀ / sqrt(64)) V per (b,h).
// 256 threads = 8 waves; each wave owns one 16-query tile. K chunk (32 keys)
// and pre-transposed V chunk stream in via async DMA, double-buffered.
// 1/sqrt(DK) folded into the Q fragment; row reductions via DPP ROW_ROR.
// ---------------------------------------------------------------------------
__global__ __launch_bounds__(256) void attention_wmma(
    const __bf16* __restrict__ Q, const __bf16* __restrict__ Kg,
    const __bf16* __restrict__ Vtg, __bf16* __restrict__ O) {
  __shared__ __bf16 Ks[2][32][64];     // [key][d]  (B frag for QKᵀ)
  __shared__ __bf16 Vt[2][64][32];     // [d][key]  (B frag for PV)
  __shared__ __bf16 Ps[8][16][34];     // per-wave P staging [row][key]

  const int tid  = threadIdx.x;
  const int lane = tid & 31;
  const int wave = tid >> 5;
  const int half = lane >> 4;
  const int lr   = lane & 15;

  const int bh = blockIdx.x >> 3;  // 8 query-slabs (128 rows) per head
  const int qt = blockIdx.x & 7;
  const int b  = bh >> 4;          // H_ == 16
  const int h  = bh & 15;
  const int qrow0 = qt * 128 + wave * 16;
  const size_t NT = (size_t)B_ * S_;  // token count (Vt column dim)

  // Stage one 32-key chunk of K and V (1 async 16B copy each per thread).
  const int kkey = tid >> 3, kcs = tid & 7;  // K: 32 keys x 8 segs of 8 elems
  const int vdd  = tid >> 2, vcs = tid & 3;  // V: 64 d-rows x 4 segs of 8 keys
  auto stage = [&](int buf, int kb0) {
    async_copy16(&Ks[buf][kkey][kcs * 8],
                 &Kg[((size_t)(b * S_ + kb0 + kkey)) * D_ + h * DK_ + kcs * 8]);
    async_copy16(&Vt[buf][vdd][vcs * 8],
                 &Vtg[((size_t)(h * DK_ + vdd)) * NT + b * S_ + kb0 + vcs * 8]);
  };

  // Q fragments (two 16x32 A-frags covering DK=64), loaded once,
  // pre-scaled by 1/sqrt(DK) so scores come out of WMMA already scaled.
  v16bf qf[2];
  {
    const size_t qbase = ((size_t)(b * S_ + qrow0 + lr)) * D_ + (size_t)h * DK_;
#pragma unroll
    for (int dc = 0; dc < 2; ++dc) {
#pragma unroll
      for (int v = 0; v < 8; ++v) {
        const int kb = ((v < 4) ? 2 * v : 16 + 2 * (v - 4)) + 8 * half;
        qf[dc][2 * v]     = (__bf16)(0.125f * (float)Q[qbase + dc * 32 + kb]);
        qf[dc][2 * v + 1] =
            (__bf16)(0.125f * (float)Q[qbase + dc * 32 + kb + 1]);
      }
    }
  }

  v8f o[4];
#pragma unroll
  for (int dt = 0; dt < 4; ++dt) o[dt] = (v8f)0.0f;
  float m[8], l[8];
#pragma unroll
  for (int v = 0; v < 8; ++v) { m[v] = -3.0e38f; l[v] = 0.0f; }

  const int nchunk = S_ / 32;
  stage(0, 0);
  for (int c = 0; c < nchunk; ++c) {
    const int buf = c & 1;
    wait_async0();
    __syncthreads();
    if (c + 1 < nchunk) stage(buf ^ 1, (c + 1) * 32);

    // --- scores: two 16x16 key tiles, K-dim = DK split into 2x32
    v8f sc0 = (v8f)0.0f, sc1 = (v8f)0.0f;
#pragma unroll
    for (int dc = 0; dc < 2; ++dc) {
      v16bf bk0, bk1;
#pragma unroll
      for (int v = 0; v < 8; ++v) {
        const int kk = dc * 32 + ((v < 4) ? 2 * v : 16 + 2 * (v - 4)) + 8 * half;  // d index
        bk0[2 * v]     = Ks[buf][lr][kk];
        bk0[2 * v + 1] = Ks[buf][lr][kk + 1];
        bk1[2 * v]     = Ks[buf][16 + lr][kk];
        bk1[2 * v + 1] = Ks[buf][16 + lr][kk + 1];
      }
      sc0 = __builtin_amdgcn_wmma_f32_16x16x32_bf16(
          false, qf[dc], false, bk0, (short)0, sc0, false, false);
      sc1 = __builtin_amdgcn_wmma_f32_16x16x32_bf16(
          false, qf[dc], false, bk1, (short)0, sc1, false, false);
    }

    // --- online softmax (rows = 16-lane halves; DPP rotate-allreduce)
    float alpha[8], pr0[8], pr1[8];
#pragma unroll
    for (int v = 0; v < 8; ++v) {
      const float mx = rowmax16(fmaxf(sc0[v], sc1[v]));
      const float mn = fmaxf(m[v], mx);
      alpha[v] = __expf(m[v] - mn);
      m[v] = mn;
      pr0[v] = __expf(sc0[v] - mn);
      pr1[v] = __expf(sc1[v] - mn);
      const float rs = rowsum16(pr0[v] + pr1[v]);
      l[v] = l[v] * alpha[v] + rs;
    }
#pragma unroll
    for (int dt = 0; dt < 4; ++dt)
#pragma unroll
      for (int v = 0; v < 8; ++v) o[dt][v] *= alpha[v];

    // --- P (C layout) -> LDS -> A-layout bf16 fragment (wave-internal)
#pragma unroll
    for (int v = 0; v < 8; ++v) {
      const int prow = v + 8 * half;
      Ps[wave][prow][lr]      = (__bf16)pr0[v];
      Ps[wave][prow][16 + lr] = (__bf16)pr1[v];
    }
    v16bf pf;
#pragma unroll
    for (int v = 0; v < 8; ++v) {
      const int kb = ((v < 4) ? 2 * v : 16 + 2 * (v - 4)) + 8 * half;
      pf[2 * v]     = Ps[wave][lr][kb];
      pf[2 * v + 1] = Ps[wave][lr][kb + 1];
    }

    // --- O += P @ V : four 16x16 d-tiles, K-dim = 32 keys
#pragma unroll
    for (int dt = 0; dt < 4; ++dt) {
      v16bf bv;
#pragma unroll
      for (int v = 0; v < 8; ++v) {
        const int kk = ((v < 4) ? 2 * v : 16 + 2 * (v - 4)) + 8 * half;  // key index
        bv[2 * v]     = Vt[buf][dt * 16 + lr][kk];
        bv[2 * v + 1] = Vt[buf][dt * 16 + lr][kk + 1];
      }
      o[dt] = __builtin_amdgcn_wmma_f32_16x16x32_bf16(
          false, pf, false, bv, (short)0, o[dt], false, false);
    }
    __syncthreads();  // done reading buf before overwrite
  }

  // --- normalize and store bf16 attn output [B,S,D] (head-major columns)
#pragma unroll
  for (int dt = 0; dt < 4; ++dt) {
#pragma unroll
    for (int v = 0; v < 8; ++v) {
      const int row = qrow0 + v + 8 * half;
      const float val = o[dt][v] / l[v];
      O[((size_t)(b * S_ + row)) * D_ + (size_t)h * DK_ + dt * 16 + lr] =
          (__bf16)val;
    }
  }
}

// ---------------------------------------------------------------------------
// LayerNorm over last dim (D=1024). One block per row, 256 threads.
// ---------------------------------------------------------------------------
template <bool OF, bool OB>
__global__ __launch_bounds__(256) void layernorm_kernel(
    const float* __restrict__ X, const float* __restrict__ g,
    const float* __restrict__ bb, float* __restrict__ outF,
    __bf16* __restrict__ outB) {
  __shared__ float redS[8], redS2[8];
  const int tid  = threadIdx.x;
  const int lane = tid & 31;
  const int wave = tid >> 5;
  const float* x = X + (size_t)blockIdx.x * D_;

  float s = 0.0f, s2 = 0.0f;
  for (int i = tid; i < D_; i += 256) {
    const float v = x[i];
    s += v;
    s2 += v * v;
  }
#pragma unroll
  for (int off = 16; off > 0; off >>= 1) {
    s  += __shfl_xor(s, off);
    s2 += __shfl_xor(s2, off);
  }
  if (lane == 0) { redS[wave] = s; redS2[wave] = s2; }
  __syncthreads();
  float ts = 0.0f, ts2 = 0.0f;
#pragma unroll
  for (int w = 0; w < 8; ++w) { ts += redS[w]; ts2 += redS2[w]; }
  const float mean = ts * (1.0f / D_);
  const float var  = ts2 * (1.0f / D_) - mean * mean;
  const float inv  = rsqrtf(var + EPS_);

  float* of = outF + (size_t)blockIdx.x * D_;
  __bf16* ob = outB + (size_t)blockIdx.x * D_;
  for (int i = tid; i < D_; i += 256) {
    const float y = (x[i] - mean) * inv * g[i] + bb[i];
    if constexpr (OF) of[i] = y;
    if constexpr (OB) ob[i] = (__bf16)y;
  }
}

// ---------------------------------------------------------------------------
// Host orchestration
// ---------------------------------------------------------------------------
#define MB (1ull << 20)

extern "C" void kernel_launch(void* const* d_in, const int* in_sizes, int n_in,
                              void* d_out, int out_size, void* d_ws,
                              size_t ws_size, hipStream_t stream) {
  (void)in_sizes; (void)n_in; (void)out_size; (void)ws_size;

  const float* x     = (const float*)d_in[0];
  const float* Wq    = (const float*)d_in[1];
  const float* bq    = (const float*)d_in[2];
  const float* Wk    = (const float*)d_in[3];
  const float* bk    = (const float*)d_in[4];
  const float* Wv    = (const float*)d_in[5];
  const float* bv    = (const float*)d_in[6];
  const float* Wo    = (const float*)d_in[7];
  const float* bo    = (const float*)d_in[8];
  const float* g1    = (const float*)d_in[9];
  const float* b1    = (const float*)d_in[10];
  const float* W_ff1 = (const float*)d_in[11];
  const float* b_ff1 = (const float*)d_in[12];
  const float* W_ff2 = (const float*)d_in[13];
  const float* b_ff2 = (const float*)d_in[14];
  const float* g2    = (const float*)d_in[15];
  const float* b2    = (const float*)d_in[16];
  float* out = (float*)d_out;

  char* w = (char*)d_ws;
  __bf16* xb    = (__bf16*)(w + 0 * MB);    // 8MB  [B*S, D]  (reused: attn)
  __bf16* attnb = xb;                        // alias: xb dead after QKV GEMMs
  __bf16* WqT   = (__bf16*)(w + 8 * MB);    // 2MB  [D][D] transposed
  __bf16* WkT   = (__bf16*)(w + 10 * MB);   // 2MB
  __bf16* WvT   = (__bf16*)(w + 12 * MB);   // 2MB
  __bf16* WoT   = (__bf16*)(w + 14 * MB);   // 2MB
  __bf16* W1T   = (__bf16*)(w + 16 * MB);   // 8MB  [DFF][D]
  __bf16* W2T   = (__bf16*)(w + 24 * MB);   // 8MB  [D][DFF]
  __bf16* Qb    = (__bf16*)(w + 32 * MB);   // 8MB  [B*S, D]
  __bf16* Kb    = (__bf16*)(w + 40 * MB);   // 8MB  [B*S, D]
  __bf16* Vtb   = (__bf16*)(w + 48 * MB);   // 8MB  [D][B*S] (transposed!)
  float*  x1f   = (float*) (w + 56 * MB);   // 16MB attn proj + residual
  float*  ff2f  = x1f;                       // alias: x1f dead after LN1
  float*  xln1f = (float*) (w + 72 * MB);   // 16MB LN1 out (fp32 residual)
  __bf16* xln1b = (__bf16*)(w + 88 * MB);   // 8MB  LN1 out (bf16 FF input)
  __bf16* hb    = (__bf16*)(w + 96 * MB);   // 32MB relu(FF1) [B*S, DFF]

  const int M = B_ * S_;  // 4096 token rows

  // 1) converts: x straight; all weights transposed to [N][K]
  {
    const int thr = 256, blks = 2048;
    f32_to_bf16_kernel<<<blks, thr, 0, stream>>>(x, xb, M * D_);
    f32_to_bf16_T_kernel<<<blks, thr, 0, stream>>>(Wq, WqT, D_, D_);
    f32_to_bf16_T_kernel<<<blks, thr, 0, stream>>>(Wk, WkT, D_, D_);
    f32_to_bf16_T_kernel<<<blks, thr, 0, stream>>>(Wv, WvT, D_, D_);
    f32_to_bf16_T_kernel<<<blks, thr, 0, stream>>>(Wo, WoT, D_, D_);
    f32_to_bf16_T_kernel<<<blks, thr, 0, stream>>>(W_ff1, W1T, D_, DFF_);
    f32_to_bf16_T_kernel<<<blks, thr, 0, stream>>>(W_ff2, W2T, DFF_, D_);
  }

  // 2) Q/K projections -> bf16 [M,D]; V projection -> bf16 transposed [D,M]
  {
    dim3 grid(D_ / BN, M / BM);
    gemm_bf16_wmma<false, false, false, true, false><<<grid, 256, 0, stream>>>(
        xb, WqT, bq, nullptr, nullptr, Qb, nullptr, M, D_, D_);
    gemm_bf16_wmma<false, false, false, true, false><<<grid, 256, 0, stream>>>(
        xb, WkT, bk, nullptr, nullptr, Kb, nullptr, M, D_, D_);
    gemm_bf16_wmma<false, false, false, false, true><<<grid, 256, 0, stream>>>(
        xb, WvT, bv, nullptr, nullptr, nullptr, Vtb, M, D_, D_);
  }

  // 3) attention -> attnb (bf16 [B,S,D]); overwrites xb (xb consumed)
  attention_wmma<<<B_ * H_ * (S_ / 128), 256, 0, stream>>>(Qb, Kb, Vtb, attnb);

  // 4) output projection + residual x -> x1f (fp32)
  {
    dim3 grid(D_ / BN, M / BM);
    gemm_bf16_wmma<false, true, true, false, false><<<grid, 256, 0, stream>>>(
        attnb, WoT, bo, x, x1f, nullptr, nullptr, M, D_, D_);
  }

  // 5) LayerNorm 1 -> xln1f (fp32 residual) + xln1b (bf16 FF input)
  layernorm_kernel<true, true><<<M, 256, 0, stream>>>(x1f, g1, b1, xln1f,
                                                      xln1b);

  // 6) FF1: relu([M,D] x [D,DFF]) -> hb (bf16)
  {
    dim3 grid(DFF_ / BN, M / BM);
    gemm_bf16_wmma<true, false, false, true, false><<<grid, 256, 0, stream>>>(
        xln1b, W1T, b_ff1, nullptr, nullptr, hb, nullptr, M, DFF_, D_);
  }

  // 7) FF2: [M,DFF] x [DFF,D] + residual xln1f -> ff2f (fp32)
  {
    dim3 grid(D_ / BN, M / BM);
    gemm_bf16_wmma<false, true, true, false, false><<<grid, 256, 0, stream>>>(
        hb, W2T, b_ff2, xln1f, ff2f, nullptr, nullptr, M, D_, DFF_);
  }

  // 8) LayerNorm 2 -> final fp32 output
  layernorm_kernel<true, false><<<M, 256, 0, stream>>>(ff2f, g2, b2, out,
                                                       nullptr);
}
